// DSOAgent_60859686584838
// MI455X (gfx1250) — matrix operand order repaired and checked by reading, not compiled, not run on hardware
//
#include <hip/hip_runtime.h>

// ---------------- static problem sizes (mirror reference) ----------------
#define T_STEPS 128
#define NB      4096
#define IN_SZ   135
#define IN_PAD  160   // pad K of Wih0 / x so fragment loads are 16B/32B aligned
#define HSZ     128
#define PSZ     64
#define GSZ     512   // 4*H

typedef __attribute__((ext_vector_type(16))) __bf16 v16bf;
typedef __attribute__((ext_vector_type(8)))  __bf16 v8bf;
typedef __attribute__((ext_vector_type(8)))  float  v8f;

__device__ __forceinline__ __bf16 f2bf(float f) {
  unsigned u = __builtin_bit_cast(unsigned, f);
  unsigned r = u + 0x7FFFu + ((u >> 16) & 1u);   // round-to-nearest-even
  unsigned short h = (unsigned short)(r >> 16);
  return __builtin_bit_cast(__bf16, h);
}
__device__ __forceinline__ __bf16 bf_zero() {
  unsigned short z = 0; return __builtin_bit_cast(__bf16, z);
}
__device__ __forceinline__ float sigmoidf_(float x) { return 1.0f / (1.0f + __expf(-x)); }

// A fragment (16x32 bf16, M x K) in contiguous k order (same as load_b_frag):
// lanes 0-15: row M=lane, elems 0..15 = K kb..kb+15, kb=0
// lanes 16-31: same with kb=16  (K 16..31)
__device__ __forceinline__ v16bf load_a_frag(const __bf16* base, int ld, int lane, int kc) {
  const int m  = lane & 15;
  const int kb = kc * 32 + ((lane >> 4) << 4);
  const __bf16* p = base + m * ld + kb;
  v8bf lo = *(const v8bf*)(p);          // 16B ds/global load
  v8bf hi = *(const v8bf*)(p + 8);      // 16B ds/global load
  v16bf r;
#pragma unroll
  for (int i = 0; i < 8; ++i) { r[i] = lo[i]; r[i + 8] = hi[i]; }
  return r;
}

// B fragment (32x16 bf16, K x N) with B[k][n] = W[n][k] (W row-major, stride ld):
// lane l: col n = l&15, elems 0..15 = K kc*32 + (l>=16 ? 16 : 0) + e  -> one 32B load
__device__ __forceinline__ v16bf load_b_frag(const __bf16* w, int ld, int lane, int kc) {
  const int n  = lane & 15;
  const int kb = kc * 32 + ((lane >> 4) << 4);
  return *(const v16bf*)(w + n * ld + kb);
}

#define WMMA_BF16(A, B, C) \
  __builtin_amdgcn_wmma_f32_16x16x32_bf16(false, (A), false, (B), (short)0, (C), false, false)

// ---------------- init: bf16 weight conversion + state broadcast ----------------
__global__ void dso_init_kernel(
    const float* __restrict__ Wih0, const float* __restrict__ Whh0, const float* __restrict__ Whr0,
    const float* __restrict__ Wih1, const float* __restrict__ Whh1, const float* __restrict__ Whr1,
    const float* __restrict__ h_init, const float* __restrict__ c_init,
    __bf16* __restrict__ Wih0b, __bf16* __restrict__ Whh0b, __bf16* __restrict__ Whr0b,
    __bf16* __restrict__ Wih1b, __bf16* __restrict__ Whh1b, __bf16* __restrict__ Whr1b,
    __bf16* __restrict__ h0, __bf16* __restrict__ h1,
    float* __restrict__ c0, float* __restrict__ c1)
{
  const int C0 = GSZ * IN_PAD;
  const int C1 = C0 + GSZ * PSZ;
  const int C2 = C1 + GSZ * PSZ;
  const int C3 = C2 + GSZ * PSZ;
  const int C4 = C3 + PSZ * HSZ;
  const int C5 = C4 + PSZ * HSZ;
  const int C6 = C5 + NB * PSZ;
  const int C7 = C6 + NB * PSZ;
  const int C8 = C7 + NB * HSZ;
  const int C9 = C8 + NB * HSZ;
  for (int i = blockIdx.x * blockDim.x + threadIdx.x; i < C9; i += gridDim.x * blockDim.x) {
    if (i < C0) {
      int g = i / IN_PAD, k = i - g * IN_PAD;
      Wih0b[i] = (k < IN_SZ) ? f2bf(Wih0[g * IN_SZ + k]) : bf_zero();
    } else if (i < C1) { int j = i - C0; Whh0b[j] = f2bf(Whh0[j]); }
    else if (i < C2)   { int j = i - C1; Wih1b[j] = f2bf(Wih1[j]); }
    else if (i < C3)   { int j = i - C2; Whh1b[j] = f2bf(Whh1[j]); }
    else if (i < C4)   { int j = i - C3; Whr0b[j] = f2bf(Whr0[j]); }
    else if (i < C5)   { int j = i - C4; Whr1b[j] = f2bf(Whr1[j]); }
    else if (i < C6)   { int j = i - C5; h0[j] = f2bf(h_init[j & (PSZ - 1)]); }
    else if (i < C7)   { int j = i - C6; h1[j] = f2bf(h_init[PSZ + (j & (PSZ - 1))]); }
    else if (i < C8)   { int j = i - C7; c0[j] = c_init[j & (HSZ - 1)]; }
    else               { int j = i - C8; c1[j] = c_init[HSZ + (j & (HSZ - 1))]; }
  }
}

// ---------------- one decode step: 16 batch rows / block, 8 waves ----------------
__global__ __launch_bounds__(256) void dso_step_kernel(
    int t,
    const float* __restrict__ xall,
    const float* __restrict__ b0, const float* __restrict__ b1,
    const __bf16* __restrict__ Wih0b, const __bf16* __restrict__ Whh0b,
    const __bf16* __restrict__ Whr0b,
    const __bf16* __restrict__ Wih1b, const __bf16* __restrict__ Whh1b,
    const __bf16* __restrict__ Whr1b,
    __bf16* __restrict__ h0, __bf16* __restrict__ h1,
    float* __restrict__ c0, float* __restrict__ c1,
    const int* __restrict__ tokens,
    float* __restrict__ out_ent, float* __restrict__ out_lp)
{
  const int nb   = blockIdx.x * 16;
  const int tid  = threadIdx.x;
  const int wave = tid >> 5;
  const int lane = tid & 31;

  __shared__ alignas(32) __bf16 sX [16][IN_PAD];  // x tile, bf16, K-padded
  __shared__ alignas(32) __bf16 sH [16][PSZ];     // staged h0 then h1 tile
  __shared__ alignas(32) __bf16 sS [16][HSZ];     // sigmoid(o)*tanh(c) tile
  __shared__ alignas(32) __bf16 sHn[16][PSZ];     // h0_new (layer0 projection)
  __shared__ float sLg[16][PSZ];                  // layer1 projection (pre-prior logits)

  const float* x = xall + (size_t)t * NB * IN_SZ;

  // speculative prefetch of next step's rows -> global_prefetch_b8
  if ((t + 1 < T_STEPS) && tid < 16)
    __builtin_prefetch(x + (size_t)NB * IN_SZ + (size_t)(nb + tid) * IN_SZ, 0, 1);

  for (int i = tid; i < 16 * IN_PAD; i += 256) {
    int r = i / IN_PAD, k = i - r * IN_PAD;
    sX[r][k] = (k < IN_SZ) ? f2bf(x[(size_t)(nb + r) * IN_SZ + k]) : bf_zero();
  }
  for (int i = tid; i < 16 * PSZ; i += 256)
    sH[i >> 6][i & 63] = h0[(size_t)(nb + (i >> 6)) * PSZ + (i & 63)];
  __syncthreads();

  v8f acc[4];

  // -------- layer 0 gates: wave w -> cols 16w..16w+15 of each of i,f,g,o --------
#pragma unroll
  for (int j = 0; j < 4; ++j) {
    const int ct = wave + 8 * j;                  // column tile 0..31 of the 512 gates
    const float bias = b0[ct * 16 + (lane & 15)];
    v8f a = {bias, bias, bias, bias, bias, bias, bias, bias};
#pragma unroll
    for (int kc = 0; kc < 5; ++kc)                // K = 160 (135 padded)
      a = WMMA_BF16(load_a_frag(&sX[0][0], IN_PAD, lane, kc),
                    load_b_frag(Wih0b + (size_t)ct * 16 * IN_PAD, IN_PAD, lane, kc), a);
#pragma unroll
    for (int kc = 0; kc < 2; ++kc)                // K = 64 (h0)
      a = WMMA_BF16(load_a_frag(&sH[0][0], PSZ, lane, kc),
                    load_b_frag(Whh0b + (size_t)ct * 16 * PSZ, PSZ, lane, kc), a);
    acc[j] = a;
  }
  // -------- layer 0 cell (register-resident on WMMA C layout) --------
  {
    const int hcol  = wave * 16 + (lane & 15);
    const int rbase = (lane >> 4) * 8;
#pragma unroll
    for (int v = 0; v < 8; ++v) {
      const int r = rbase + v;
      const size_t ci = (size_t)(nb + r) * HSZ + hcol;
      const float iv = sigmoidf_(acc[0][v]);
      const float fv = sigmoidf_(acc[1][v]);
      const float gv = tanhf(acc[2][v]);
      const float ov = sigmoidf_(acc[3][v]);
      const float cn = fv * c0[ci] + iv * gv;
      c0[ci] = cn;
      sS[r][hcol] = f2bf(ov * tanhf(cn));
    }
  }
  __syncthreads();

  // -------- layer 0 projection: h0n = sS[16x128] @ Whr0^T --------
  if (wave < 4) {
    v8f a = {0.f, 0.f, 0.f, 0.f, 0.f, 0.f, 0.f, 0.f};
#pragma unroll
    for (int kc = 0; kc < 4; ++kc)
      a = WMMA_BF16(load_a_frag(&sS[0][0], HSZ, lane, kc),
                    load_b_frag(Whr0b + (size_t)wave * 16 * HSZ, HSZ, lane, kc), a);
    const int col = wave * 16 + (lane & 15);
    const int rbase = (lane >> 4) * 8;
#pragma unroll
    for (int v = 0; v < 8; ++v) sHn[rbase + v][col] = f2bf(a[v]);
  }
  __syncthreads();

  // commit h0, stage h1
  for (int i = tid; i < 16 * PSZ; i += 256) {
    const int r = i >> 6, cix = i & 63;
    h0[(size_t)(nb + r) * PSZ + cix] = sHn[r][cix];
    sH[r][cix] = h1[(size_t)(nb + r) * PSZ + cix];
  }
  __syncthreads();

  // -------- layer 1 gates --------
#pragma unroll
  for (int j = 0; j < 4; ++j) {
    const int ct = wave + 8 * j;
    const float bias = b1[ct * 16 + (lane & 15)];
    v8f a = {bias, bias, bias, bias, bias, bias, bias, bias};
#pragma unroll
    for (int kc = 0; kc < 2; ++kc)                // K = 64 (h0n)
      a = WMMA_BF16(load_a_frag(&sHn[0][0], PSZ, lane, kc),
                    load_b_frag(Wih1b + (size_t)ct * 16 * PSZ, PSZ, lane, kc), a);
#pragma unroll
    for (int kc = 0; kc < 2; ++kc)                // K = 64 (h1)
      a = WMMA_BF16(load_a_frag(&sH[0][0], PSZ, lane, kc),
                    load_b_frag(Whh1b + (size_t)ct * 16 * PSZ, PSZ, lane, kc), a);
    acc[j] = a;
  }
  // -------- layer 1 cell --------
  {
    const int hcol  = wave * 16 + (lane & 15);
    const int rbase = (lane >> 4) * 8;
#pragma unroll
    for (int v = 0; v < 8; ++v) {
      const int r = rbase + v;
      const size_t ci = (size_t)(nb + r) * HSZ + hcol;
      const float iv = sigmoidf_(acc[0][v]);
      const float fv = sigmoidf_(acc[1][v]);
      const float gv = tanhf(acc[2][v]);
      const float ov = sigmoidf_(acc[3][v]);
      const float cn = fv * c1[ci] + iv * gv;
      c1[ci] = cn;
      sS[r][hcol] = f2bf(ov * tanhf(cn));
    }
  }
  __syncthreads();

  // -------- layer 1 projection -> logits (f32 in LDS) --------
  if (wave < 4) {
    v8f a = {0.f, 0.f, 0.f, 0.f, 0.f, 0.f, 0.f, 0.f};
#pragma unroll
    for (int kc = 0; kc < 4; ++kc)
      a = WMMA_BF16(load_a_frag(&sS[0][0], HSZ, lane, kc),
                    load_b_frag(Whr1b + (size_t)wave * 16 * HSZ, HSZ, lane, kc), a);
    const int col = wave * 16 + (lane & 15);
    const int rbase = (lane >> 4) * 8;
#pragma unroll
    for (int v = 0; v < 8; ++v) sLg[rbase + v][col] = a[v];
  }
  __syncthreads();

  // commit h1
  for (int i = tid; i < 16 * PSZ; i += 256)
    h1[(size_t)(nb + (i >> 6)) * PSZ + (i & 63)] = f2bf(sLg[i >> 6][i & 63]);

  // -------- length prior + softmax + entropy + token log-prob --------
  if (tid < 16) {
    const int n = nb + tid;
    const float tf = (float)t;
    const float pshort = (tf < 64.f) ? -((64.f - tf) * (64.f - tf)) * 0.0625f : 0.f;
    const float plong  = (tf > 64.f) ? -((tf - 64.f) * (tf - 64.f)) * 0.0625f : 0.f;

    float mx = -3.0e38f;
    for (int j2 = 0; j2 < PSZ; ++j2) {
      const float pr = (j2 < 32) ? pshort : ((j2 >= 48) ? plong : 0.f);
      mx = fmaxf(mx, sLg[tid][j2] + pr);
    }
    float s = 0.f;
    for (int j2 = 0; j2 < PSZ; ++j2) {
      const float pr = (j2 < 32) ? pshort : ((j2 >= 48) ? plong : 0.f);
      s += __expf(sLg[tid][j2] + pr - mx);
    }
    const float inv = 1.f / s;
    const float renorm = 1.f / (1.f + 64.f * 1e-10f);
    const int tok = tokens[(size_t)n * T_STEPS + t];
    float ent = 0.f, lpt = 0.f;
    for (int j2 = 0; j2 < PSZ; ++j2) {
      const float pr = (j2 < 32) ? pshort : ((j2 >= 48) ? plong : 0.f);
      const float p  = (__expf(sLg[tid][j2] + pr - mx) * inv + 1e-10f) * renorm;
      const float lp = __logf(p);
      ent -= p * lp;
      if (j2 == tok) lpt = lp;
    }
    out_ent[(size_t)n * T_STEPS + t] = ent;
    out_lp [(size_t)n * T_STEPS + t] = lpt;
  }
}

extern "C" void kernel_launch(void* const* d_in, const int* in_sizes, int n_in,
                              void* d_out, int out_size, void* d_ws, size_t ws_size,
                              hipStream_t stream) {
  (void)in_sizes; (void)n_in; (void)out_size; (void)ws_size;
  const float* xall   = (const float*)d_in[0];
  const float* Wih0   = (const float*)d_in[1];
  const float* Whh0   = (const float*)d_in[2];
  const float* b0     = (const float*)d_in[3];
  const float* Whr0   = (const float*)d_in[4];
  const float* Wih1   = (const float*)d_in[5];
  const float* Whh1   = (const float*)d_in[6];
  const float* b1     = (const float*)d_in[7];
  const float* Whr1   = (const float*)d_in[8];
  const float* h_init = (const float*)d_in[9];
  const float* c_init = (const float*)d_in[10];
  const int*   tokens = (const int*)d_in[11];

  char* ws = (char*)d_ws;
  size_t off = 0;
  auto take = [&](size_t bytes) -> char* {
    char* p = ws + off;
    off = (off + bytes + 255) & ~(size_t)255;
    return p;
  };
  __bf16* Wih0b = (__bf16*)take((size_t)GSZ * IN_PAD * 2);
  __bf16* Whh0b = (__bf16*)take((size_t)GSZ * PSZ * 2);
  __bf16* Wih1b = (__bf16*)take((size_t)GSZ * PSZ * 2);
  __bf16* Whh1b = (__bf16*)take((size_t)GSZ * PSZ * 2);
  __bf16* Whr0b = (__bf16*)take((size_t)PSZ * HSZ * 2);
  __bf16* Whr1b = (__bf16*)take((size_t)PSZ * HSZ * 2);
  __bf16* h0    = (__bf16*)take((size_t)NB * PSZ * 2);
  __bf16* h1    = (__bf16*)take((size_t)NB * PSZ * 2);
  float*  c0    = (float*) take((size_t)NB * HSZ * 4);
  float*  c1    = (float*) take((size_t)NB * HSZ * 4);

  dso_init_kernel<<<2048, 256, 0, stream>>>(
      Wih0, Whh0, Whr0, Wih1, Whh1, Whr1, h_init, c_init,
      Wih0b, Whh0b, Whr0b, Wih1b, Whh1b, Whr1b, h0, h1, c0, c1);

  float* out_ent = (float*)d_out;
  float* out_lp  = out_ent + (size_t)NB * T_STEPS;
  for (int t = 0; t < T_STEPS; ++t) {
    dso_step_kernel<<<NB / 16, 256, 0, stream>>>(
        t, xall, b0, b1, Wih0b, Whh0b, Whr0b, Wih1b, Whh1b, Whr1b,
        h0, h1, c0, c1, tokens, out_ent, out_lp);
  }
}
